// MultiQueryAttention_80814104641917
// MI455X (gfx1250) — matrix operand order repaired and checked by reading, not compiled, not run on hardware
//
#include <hip/hip_runtime.h>
#include <hip/hip_bf16.h>

// -------- fixed problem shape (from reference) --------
constexpr int Bb   = 2;
constexpr int Qq   = 2048;
constexpr int KVn  = 2048;
constexpr int En   = 1024;
constexpr int Hn   = 16;
constexpr int KVHn = 4;
constexpr int HDn  = 64;

typedef __attribute__((ext_vector_type(16))) __bf16 bf16x16;
typedef __attribute__((ext_vector_type(8)))  float  f32x8;
typedef __attribute__((ext_vector_type(2)))  float  f32x2;

union FragB16 { bf16x16 v; uint4 u[2]; };

// packed f32x2 -> bf16x2 (RNE) in one VALU op
__device__ __forceinline__ unsigned int f2bf_pk(float a, float b) {
  unsigned int r;
  asm("v_cvt_pk_bf16_f32 %0, %1, %2" : "=v"(r) : "v"(a), "v"(b));
  return r;   // [15:0] = bf16(a), [31:16] = bf16(b)
}
__device__ __forceinline__ unsigned short f2bf(float f) {
  unsigned int u = __float_as_uint(f);
  u += 0x7FFFu + ((u >> 16) & 1u);          // round-to-nearest-even
  return (unsigned short)(u >> 16);
}
__device__ __forceinline__ float bf2f(unsigned short h) {
  return __uint_as_float(((unsigned int)h) << 16);
}

// A fragment (16x32 bf16, M x K): lane holds row M=lane%16.
// v0-3 hold K = hi*8 .. hi*8+7 ; v4-7 hold K = 16+hi*8 .. +7   (ISA 7.12.2)
__device__ __forceinline__ bf16x16 load_frag_a(const unsigned short* row, int hi) {
  FragB16 f;
  f.u[0] = *(const uint4*)(row + hi * 8);
  f.u[1] = *(const uint4*)(row + 16 + hi * 8);
  return f.v;
}
// B fragment (32x16 bf16, K x N): lane holds column N=lane%16.
// v0-3 hold K = hi*8 .. hi*8+7 ; v4-7 hold K = 16+hi*8 .. +7 (same as A)
__device__ __forceinline__ bf16x16 load_frag_b(const unsigned short* col, int hi) {
  FragB16 f;
  f.u[0] = *(const uint4*)(col + hi * 8);
  f.u[1] = *(const uint4*)(col + 16 + hi * 8);
  return f.v;
}
__device__ __forceinline__ f32x8 wmma_bf16(bf16x16 a, bf16x16 b, f32x8 c) {
  return __builtin_amdgcn_wmma_f32_16x16x32_bf16(false, a, false, b, (short)0, c,
                                                 false, false);
}

// ============================================================================
// Generic WMMA GEMM:  C[M,N] = A[M,K] @ W[K,N] (+bias) (+addmat)
// A fp32 (packed-converted to bf16 in LDS) or already bf16. Out fp32 or bf16.
// Block tile 128x64, K-step 32, 8 waves (4x2), 2x2 WMMA tiles per wave.
// Double-buffered LDS: one barrier per K-step, loads overlap WMMA.
// ============================================================================
__global__ __launch_bounds__(256) void gemm_wmma_bf16(
    const float* __restrict__ Af, const unsigned short* __restrict__ Ab,
    const float* __restrict__ W, const float* __restrict__ bias,
    const float* __restrict__ addmat,
    float* __restrict__ Cf, unsigned short* __restrict__ Cb,
    int M, int N, int K)
{
  __shared__ unsigned short Alds[2][128][40];   // 32 cols + 8 pad (bank spread)
  __shared__ unsigned short Blds[2][64][40];    // stored as [n][k]

  const int t = threadIdx.x;
  const int lane = t & 31, w = t >> 5;
  const int l16 = lane & 15, hi = lane >> 4;
  const int m0 = blockIdx.y * 128, n0 = blockIdx.x * 64;
  const int wm = w & 3, wn = w >> 2;

  f32x8 acc00 = {0,0,0,0,0,0,0,0};
  f32x8 acc01 = acc00, acc10 = acc00, acc11 = acc00;

  const int ar = t >> 1, ac0 = (t & 1) * 16;   // A tile: 128 rows, 2 thr/row
  const int bkr = t >> 3, bc0 = (t & 7) * 8;   // W tile: 32 rows, 8 thr/row

  auto stage = [&](int buf, int k0) {
    if (Af) {
      const float* src = Af + (size_t)(m0 + ar) * K + (k0 + ac0);
      unsigned int* dst = (unsigned int*)&Alds[buf][ar][ac0];
      float4 f0 = *(const float4*)(src + 0);
      float4 f1 = *(const float4*)(src + 4);
      float4 f2 = *(const float4*)(src + 8);
      float4 f3 = *(const float4*)(src + 12);
      dst[0] = f2bf_pk(f0.x, f0.y);  dst[1] = f2bf_pk(f0.z, f0.w);
      dst[2] = f2bf_pk(f1.x, f1.y);  dst[3] = f2bf_pk(f1.z, f1.w);
      dst[4] = f2bf_pk(f2.x, f2.y);  dst[5] = f2bf_pk(f2.z, f2.w);
      dst[6] = f2bf_pk(f3.x, f3.y);  dst[7] = f2bf_pk(f3.z, f3.w);
    } else {
      const unsigned short* src = Ab + (size_t)(m0 + ar) * K + (k0 + ac0);
      *(uint4*)&Alds[buf][ar][ac0]     = *(const uint4*)src;
      *(uint4*)&Alds[buf][ar][ac0 + 8] = *(const uint4*)(src + 8);
    }
    {
      const float* src = W + (size_t)(k0 + bkr) * N + (n0 + bc0);
      float4 f0 = *(const float4*)(src + 0);
      float4 f1 = *(const float4*)(src + 4);
      const unsigned int p0 = f2bf_pk(f0.x, f0.y), p1 = f2bf_pk(f0.z, f0.w);
      const unsigned int p2 = f2bf_pk(f1.x, f1.y), p3 = f2bf_pk(f1.z, f1.w);
      Blds[buf][bc0+0][bkr] = (unsigned short)p0;
      Blds[buf][bc0+1][bkr] = (unsigned short)(p0 >> 16);
      Blds[buf][bc0+2][bkr] = (unsigned short)p1;
      Blds[buf][bc0+3][bkr] = (unsigned short)(p1 >> 16);
      Blds[buf][bc0+4][bkr] = (unsigned short)p2;
      Blds[buf][bc0+5][bkr] = (unsigned short)(p2 >> 16);
      Blds[buf][bc0+6][bkr] = (unsigned short)p3;
      Blds[buf][bc0+7][bkr] = (unsigned short)(p3 >> 16);
    }
    if (k0 + 32 < K) {   // hint tile after next into cache (global_prefetch)
      __builtin_prefetch(W + (size_t)(k0 + 32 + bkr) * N + (n0 + bc0), 0, 1);
      if (Af) __builtin_prefetch(Af + (size_t)(m0 + ar) * K + (k0 + 32 + ac0), 0, 1);
      else    __builtin_prefetch(Ab + (size_t)(m0 + ar) * K + (k0 + 32 + ac0), 0, 1);
    }
  };

  stage(0, 0);
  int cur = 0;
  for (int k0 = 0; k0 < K; k0 += 32) {
    __syncthreads();
    if (k0 + 32 < K) stage(cur ^ 1, k0 + 32);   // overlap next-tile loads w/ WMMA
    bf16x16 a0 = load_frag_a(&Alds[cur][wm*32      + l16][0], hi);
    bf16x16 a1 = load_frag_a(&Alds[cur][wm*32 + 16 + l16][0], hi);
    bf16x16 b0 = load_frag_b(&Blds[cur][wn*32      + l16][0], hi);
    bf16x16 b1 = load_frag_b(&Blds[cur][wn*32 + 16 + l16][0], hi);
    acc00 = wmma_bf16(a0, b0, acc00);
    acc01 = wmma_bf16(a0, b1, acc01);
    acc10 = wmma_bf16(a1, b0, acc10);
    acc11 = wmma_bf16(a1, b1, acc11);
    cur ^= 1;
  }

  // ---- epilogue: C layout m = v + 8*hi, n = lane%16
  auto store_tile = [&](f32x8 accv, int i, int j) {
    const int col = n0 + wn * 32 + j * 16 + l16;
    const float bv_ = bias ? bias[col] : 0.0f;
    #pragma unroll
    for (int v = 0; v < 8; ++v) {
      const int row = m0 + wm * 32 + i * 16 + v + hi * 8;
      float val = accv[v] + bv_;
      if (addmat) val += addmat[(size_t)row * N + col];
      if (Cf) Cf[(size_t)row * N + col] = val;
      else    Cb[(size_t)row * N + col] = f2bf(val);
    }
  };
  store_tile(acc00, 0, 0); store_tile(acc01, 0, 1);
  store_tile(acc10, 1, 0); store_tile(acc11, 1, 1);
}

// ============================================================================
// Kexp[b][h][kv][d] = bf16( K4[b][kv][(h/4)*64+d] + key_pos[b][kv][h*64+d] )
// ============================================================================
__global__ __launch_bounds__(256) void build_kexp(
    const unsigned short* __restrict__ K4, const float* __restrict__ kpos,
    unsigned short* __restrict__ Kexp)
{
  const size_t i = (size_t)blockIdx.x * blockDim.x + threadIdx.x;
  if (i >= (size_t)Bb * Hn * KVn * HDn) return;
  const int d  = (int)(i & (HDn - 1));
  const int kv = (int)((i >> 6)  & (KVn - 1));
  const int h  = (int)((i >> 17) & (Hn - 1));
  const int b  = (int)(i >> 21);
  const size_t rowk = (size_t)(b * KVn + kv);
  const float kval = bf2f(K4[rowk * (KVHn * HDn) + (h >> 2) * HDn + d]);
  const float pv   = kpos[rowk * En + h * HDn + d];
  Kexp[i] = f2bf(kval + pv);
}

// Vt[b][kvh][d][kv] = V4[b][kv][kvh*64+d]   (d-major so P@V B-fragments are contiguous)
__global__ __launch_bounds__(256) void build_vt(
    const unsigned short* __restrict__ V4, unsigned short* __restrict__ Vt)
{
  const size_t i = (size_t)blockIdx.x * blockDim.x + threadIdx.x;
  if (i >= (size_t)Bb * KVHn * HDn * KVn) return;
  const int kv  = (int)(i & (KVn - 1));
  const int d   = (int)((i >> 11) & (HDn - 1));
  const int kvh = (int)((i >> 17) & (KVHn - 1));
  const int b   = (int)(i >> 19);
  Vt[i] = V4[(size_t)(b * KVn + kv) * (KVHn * HDn) + kvh * HDn + d];
}

// ============================================================================
// Fused attention: one workgroup = one (b, h, 16-query strip).
// LDS: S 16x2052 f32 | P 16x2056 bf16 | Qtile | O partials  (~203 KB < 320 KB)
// S = QK^T/8 -> softmax (wave32 shuffles) -> attn written nontemporal -> O = P@V
// ============================================================================
constexpr int    S_STR  = KVn + 4;      // fp32 row stride (16B pad)
constexpr int    P_STR  = KVn + 8;      // bf16 row stride (16B pad)
constexpr int    QT_STR = HDn + 8;
constexpr size_t OFF_P  = (size_t)16 * S_STR * 4;               // 131328
constexpr size_t OFF_QT = OFF_P  + (size_t)16 * P_STR * 2;      // 197120
constexpr size_t OFF_OB = OFF_QT + (size_t)16 * QT_STR * 2;     // 199424
constexpr size_t ATTN_SMEM = OFF_OB + 2 * 16 * 64 * 4;          // 207616

__global__ __launch_bounds__(256) void attn_fused(
    const unsigned short* __restrict__ Qbf,   // [B*Q, 1024] bf16
    const unsigned short* __restrict__ Kexp,  // [B,H,KV,64] bf16
    const unsigned short* __restrict__ Vt,    // [B,KVH,64,KV] bf16
    float* __restrict__ attnp,                // [B,H,Q,KV] fp32 (final output)
    unsigned short* __restrict__ Obf)         // [B*Q, 1024] bf16
{
  extern __shared__ __align__(16) char smem[];
  float*          S  = (float*)(smem);
  unsigned short* P  = (unsigned short*)(smem + OFF_P);
  unsigned short* Qt = (unsigned short*)(smem + OFF_QT);
  float*          OB = (float*)(smem + OFF_OB);

  const int t = threadIdx.x;
  const int lane = t & 31, w = t >> 5;
  const int l16 = lane & 15, hi = lane >> 4;
  const int q0 = blockIdx.x * 16;
  const int h  = blockIdx.y;
  const int b  = blockIdx.z;

  // ---- stage 16x64 Q tile
  {
    const int r = t >> 4, c = (t & 15) * 4;
    const unsigned short* src = Qbf + (size_t)(b * Qq + q0 + r) * En + h * HDn + c;
    *(uint2*)&Qt[r * QT_STR + c] = *(const uint2*)src;
  }
  __syncthreads();

  const unsigned short* qrow = Qt + l16 * QT_STR;
  const bf16x16 aq0 = load_frag_a(qrow, hi);        // d = 0..31
  const bf16x16 aq1 = load_frag_a(qrow + 32, hi);   // d = 32..63

  // ---- S = (Q . K^T)/8 ; each wave owns a 256-wide KV window (16 tiles)
  const unsigned short* Kbase = Kexp + (size_t)(b * Hn + h) * KVn * HDn;
  for (int tk = 0; tk < 16; ++tk) {
    const int kv0 = w * 256 + tk * 16;
    const unsigned short* krow = Kbase + (size_t)(kv0 + l16) * HDn;
    const bf16x16 kb0 = load_frag_b(krow, hi);
    const bf16x16 kb1 = load_frag_b(krow + 32, hi);
    f32x8 c = {0,0,0,0,0,0,0,0};
    c = wmma_bf16(aq0, kb0, c);
    c = wmma_bf16(aq1, kb1, c);
    #pragma unroll
    for (int v = 0; v < 8; ++v)
      S[(v + hi * 8) * S_STR + kv0 + l16] = c[v] * 0.125f;
  }
  __syncthreads();

  // ---- softmax: 2 rows per wave, 2 cols per lane; attn stored nontemporal
  #pragma unroll
  for (int rr = 0; rr < 2; ++rr) {
    const int r = w * 2 + rr;
    float* Srow = S + r * S_STR;
    float mx = -3.4e38f;
    for (int c = lane * 2; c < KVn; c += 64) {
      const f32x2 s2 = *(const f32x2*)(Srow + c);
      mx = fmaxf(mx, fmaxf(s2.x, s2.y));
    }
    #pragma unroll
    for (int off = 16; off > 0; off >>= 1) mx = fmaxf(mx, __shfl_xor(mx, off, 32));
    float sum = 0.0f;
    for (int c = lane * 2; c < KVn; c += 64) {
      f32x2 s2 = *(const f32x2*)(Srow + c);
      s2.x = __expf(s2.x - mx);
      s2.y = __expf(s2.y - mx);
      *(f32x2*)(Srow + c) = s2;
      sum += s2.x + s2.y;
    }
    #pragma unroll
    for (int off = 16; off > 0; off >>= 1) sum += __shfl_xor(sum, off, 32);
    const float inv = 1.0f / sum;
    float* arow = attnp + ((size_t)(b * Hn + h) * Qq + (q0 + r)) * KVn;
    unsigned short* Prow = P + r * P_STR;
    for (int c = lane * 2; c < KVn; c += 64) {
      const f32x2 s2 = *(const f32x2*)(Srow + c);
      const f32x2 p2 = {s2.x * inv, s2.y * inv};
      __builtin_nontemporal_store(p2, (f32x2*)(arow + c));   // 8B NT store
      *(unsigned int*)&Prow[c] = f2bf_pk(p2.x, p2.y);        // packed b32 LDS store
    }
  }
  __syncthreads();

  // ---- O = P @ V : wave w -> d-tile (w&3), KV-half (w>>2)
  const int ntile = w & 3, khalf = w >> 2;
  const unsigned short* vcol =
      Vt + ((size_t)(b * KVHn + (h >> 2)) * HDn + ntile * 16 + l16) * (size_t)KVn;
  const unsigned short* prow = P + l16 * P_STR;
  f32x8 acc = {0,0,0,0,0,0,0,0};
  for (int ks = 0; ks < 32; ++ks) {
    const int kc = khalf * 1024 + ks * 32;
    const bf16x16 pa = load_frag_a(prow + kc, hi);
    const bf16x16 vb = load_frag_b(vcol + kc, hi);
    acc = wmma_bf16(pa, vb, acc);
  }
  float* ob = OB + khalf * 1024;
  #pragma unroll
  for (int v = 0; v < 8; ++v)
    ob[(v + hi * 8) * 64 + ntile * 16 + l16] = acc[v];
  __syncthreads();

  // ---- combine the two KV halves, emit bf16 attention output
  #pragma unroll
  for (int i2 = 0; i2 < 4; ++i2) {
    const int idx = t * 4 + i2;
    const int r = idx >> 6, cc = idx & 63;
    const float o = OB[r * 64 + cc] + OB[1024 + r * 64 + cc];
    Obf[(size_t)(b * Qq + q0 + r) * En + h * HDn + cc] = f2bf(o);
  }
}

// ============================================================================
// Host launch
// ============================================================================
extern "C" void kernel_launch(void* const* d_in, const int* in_sizes, int n_in,
                              void* d_out, int out_size, void* d_ws, size_t ws_size,
                              hipStream_t stream)
{
  (void)in_sizes; (void)n_in; (void)out_size; (void)ws_size;
  const float* hidden = (const float*)d_in[0];
  const float* key    = (const float*)d_in[1];
  const float* value  = (const float*)d_in[2];
  const float* qpos   = (const float*)d_in[3];
  const float* kpos   = (const float*)d_in[4];
  const float* Wq     = (const float*)d_in[5];
  const float* bq     = (const float*)d_in[6];
  const float* Wk     = (const float*)d_in[7];
  const float* bk     = (const float*)d_in[8];
  const float* Wv     = (const float*)d_in[9];
  const float* bv     = (const float*)d_in[10];
  const float* Wo     = (const float*)d_in[11];

  // workspace layout (bytes): all 256B-aligned, total 30 MB
  char* ws = (char*)d_ws;
  unsigned short* Qbf  = (unsigned short*)(ws + 0);          //  8 MB [B*Q,1024] bf16
  unsigned short* K4   = (unsigned short*)(ws + 8388608);    //  2 MB [B*KV,256] bf16
  unsigned short* V4   = (unsigned short*)(ws + 10485760);   //  2 MB
  unsigned short* Kexp = (unsigned short*)(ws + 12582912);   //  8 MB [B,H,KV,64] bf16
  unsigned short* Vt   = (unsigned short*)(ws + 20971520);   //  2 MB [B,KVH,64,KV] bf16
  unsigned short* Obf  = (unsigned short*)(ws + 23068672);   //  8 MB [B*Q,1024] bf16

  float* out  = (float*)d_out;                               // [B,Q,E]
  float* attn = out + (size_t)Bb * Qq * En;                  // [B,H,Q,KV]

  const dim3 blk(256);

  // 1) Q = hs@Wq + bq + query_pos  -> bf16
  gemm_wmma_bf16<<<dim3(En / 64, (Bb * Qq) / 128), blk, 0, stream>>>(
      hidden, nullptr, Wq, bq, qpos, nullptr, Qbf, Bb * Qq, En, En);
  // 2) K4 = key@Wk + bk -> bf16
  gemm_wmma_bf16<<<dim3((KVHn * HDn) / 64, (Bb * KVn) / 128), blk, 0, stream>>>(
      key, nullptr, Wk, bk, nullptr, nullptr, K4, Bb * KVn, KVHn * HDn, En);
  // 3) V4 = value@Wv + bv -> bf16
  gemm_wmma_bf16<<<dim3((KVHn * HDn) / 64, (Bb * KVn) / 128), blk, 0, stream>>>(
      value, nullptr, Wv, bv, nullptr, nullptr, V4, Bb * KVn, KVHn * HDn, En);
  // 4) per-head expanded K (+key_pos) and d-major V (both L2-resident)
  build_kexp<<<(Bb * Hn * KVn * HDn) / 256, blk, 0, stream>>>(K4, kpos, Kexp);
  build_vt<<<(Bb * KVHn * HDn * KVn) / 256, blk, 0, stream>>>(V4, Vt);
  // 5) fused attention (writes attn output once, nontemporal)
  attn_fused<<<dim3(Qq / 16, Hn, Bb), blk, ATTN_SMEM, stream>>>(
      Qbf, Kexp, Vt, attn, Obf);
  // 6) out = O@Wo + hidden -> fp32
  gemm_wmma_bf16<<<dim3(En / 64, (Bb * Qq) / 128), blk, 0, stream>>>(
      nullptr, Obf, Wo, nullptr, hidden, out, nullptr, Bb * Qq, En, En);
}
